// CausalSelfAttention_79748952752524
// MI455X (gfx1250) — matrix operand order repaired and checked by reading, not compiled, not run on hardware
//
#include <hip/hip_runtime.h>

typedef __attribute__((ext_vector_type(16))) _Float16 v16h;
typedef __attribute__((ext_vector_type(8)))  _Float16 v8h;
typedef __attribute__((ext_vector_type(8)))  float    v8f;

#define BB  2
#define SS  2048
#define DD  1024
#define HH  16
#define HD  64

union AFrag { v16h v; v8h h[2]; };

// ---- Fragment loaders (wave32, 16x16x32 f16 WMMA) -------------------------
// A-matrix 16x32 f16 (ISA 7.12.2): lane L (row M = L&15) holds halves i with
//   K = (i&7) + ((i&8)<<1) + ((L&16)>>1)   -> two contiguous 16-byte chunks.
// p must already point at (row, k0 + ((lane&16)>>1)).
__device__ __forceinline__ v16h frag_a(const _Float16* __restrict__ p) {
  AFrag f;
  f.h[0] = *(const v8h*)(p);
  f.h[1] = *(const v8h*)(p + 16);
  return f.v;
}

// B-matrix 32x16 f16: lane L holds column N = L&15; halves i map to
//   K = (i&7) + ((i&8)<<1) + ((L&16)>>1)  -> two contiguous 16-byte chunks. p points at (nrow, k0+((lane&16)>>1)).
__device__ __forceinline__ v16h frag_b(const _Float16* __restrict__ p) {
  AFrag f;
  f.h[0] = *(const v8h*)(p);
  f.h[1] = *(const v8h*)(p + 16);
  return f.v;
}

#define WMMA_F16(a, b, c) \
  __builtin_amdgcn_wmma_f32_16x16x32_f16(false, (a), false, (b), (short)0, (c), false, false)

// ---- fp32 -> fp16 conversion ----------------------------------------------
__global__ void f32_to_f16_kernel(const float* __restrict__ in,
                                  _Float16* __restrict__ out, int n) {
  int i = blockIdx.x * blockDim.x + threadIdx.x;
  if (i < n) out[i] = (_Float16)in[i];
}

// ---- y = A @ Bt^T + bias ; A:[M,K=DD] f16, Bt:[N,K=DD] f16, both row-major -
// Ping-pong double-buffered K loop (no register-rotation copies): buffer set A
// feeds WMMAs while set B loads, and vice versa; static tail peel.
// mode 0: f16 out, scatter to [b,h,s,hd]     (Q, K)
// mode 1: f16 out, scatter to [b,h,hd,s]     (V transposed)
// mode 2: f32 out, row-major [M,N]           (final projection -> d_out)
__global__ __launch_bounds__(32)
void gemm_xwT_kernel(const _Float16* __restrict__ A, const _Float16* __restrict__ Bt,
                     const float* __restrict__ bias, void* __restrict__ dst,
                     int N, int mode) {
  const int lane = threadIdx.x;
  const int m0 = blockIdx.x * 16;
  const int n0 = blockIdx.y * 64;
  const int arow = m0 + (lane & 15);
  const int bcol = lane & 15;

  const _Float16* pa = A + (long)arow * DD + ((lane & 16) >> 1);
  const _Float16* pb[4];
#pragma unroll
  for (int j = 0; j < 4; ++j)
    pb[j] = Bt + (long)(n0 + j * 16 + bcol) * DD + ((lane & 16) >> 1);

  v8f acc[4] = {};
  v16h aA, aB, bA[4], bB[4];

  aA = frag_a(pa);
#pragma unroll
  for (int j = 0; j < 4; ++j) bA[j] = frag_b(pb[j]);

  for (int k0 = 0; k0 < DD - 64; k0 += 64) {
    aB = frag_a(pa + k0 + 32);
#pragma unroll
    for (int j = 0; j < 4; ++j) bB[j] = frag_b(pb[j] + k0 + 32);
#pragma unroll
    for (int j = 0; j < 4; ++j) acc[j] = WMMA_F16(aA, bA[j], acc[j]);

    aA = frag_a(pa + k0 + 64);
#pragma unroll
    for (int j = 0; j < 4; ++j) bA[j] = frag_b(pb[j] + k0 + 64);
#pragma unroll
    for (int j = 0; j < 4; ++j) acc[j] = WMMA_F16(aB, bB[j], acc[j]);
  }
  // tail: buffers for k0 = DD-64 are live in set A; load DD-32 into set B
  aB = frag_a(pa + DD - 32);
#pragma unroll
  for (int j = 0; j < 4; ++j) bB[j] = frag_b(pb[j] + DD - 32);
#pragma unroll
  for (int j = 0; j < 4; ++j) acc[j] = WMMA_F16(aA, bA[j], acc[j]);
#pragma unroll
  for (int j = 0; j < 4; ++j) acc[j] = WMMA_F16(aB, bB[j], acc[j]);

  // C/D layout: VGPR r, lane L -> row m0 + r + 8*((L&16)>>4), col n0 + (L&15)
  const int mr = m0 + ((lane & 16) >> 1);
#pragma unroll
  for (int j = 0; j < 4; ++j) {
    const int c = n0 + j * 16 + (lane & 15);
    const float bv = bias[c];
#pragma unroll
    for (int r = 0; r < 8; ++r) {
      const int t = mr + r;
      const float val = acc[j][r] + bv;
      if (mode == 2) {
        ((float*)dst)[(long)t * N + c] = val;
      } else {
        const int bb = t >> 11, s = t & (SS - 1);
        const int h = c >> 6, hd = c & (HD - 1);
        long idx;
        if (mode == 0) idx = (((long)(bb * HH + h)) * SS + s) * HD + hd;
        else           idx = (((long)(bb * HH + h)) * HD + hd) * SS + s;
        ((_Float16*)dst)[idx] = (_Float16)val;
      }
    }
  }
}

// ---- scores = (Q @ K^T) * HD^-0.5, raw f32 into attn region ---------------
// HD = 64 -> exactly two K-steps; load everything, then 8 back-to-back WMMAs.
__global__ __launch_bounds__(32)
void scores_kernel(const _Float16* __restrict__ Qh, const _Float16* __restrict__ Kh,
                   float* __restrict__ attn) {
  const int lane = threadIdx.x;
  const int q16 = blockIdx.x;        // query tile (16 rows)
  const int ks  = blockIdx.y;        // key strip (64 cols)
  const int bh  = blockIdx.z;
  if (ks * 64 > q16 * 16 + 15) return;   // fully above causal diagonal: skip
  const _Float16* Qb = Qh + (long)bh * SS * HD;
  const _Float16* Kb = Kh + (long)bh * SS * HD;
  const int qrow = q16 * 16 + (lane & 15);
  const int bcol = lane & 15;

  const _Float16* pa = Qb + (long)qrow * HD + ((lane & 16) >> 1);
  v16h a0 = frag_a(pa);
  v16h a1 = frag_a(pa + 32);
  v16h b0[4], b1[4];
#pragma unroll
  for (int j = 0; j < 4; ++j) {
    const _Float16* pbj = Kb + (long)(ks * 64 + j * 16 + bcol) * HD + ((lane & 16) >> 1);
    b0[j] = frag_b(pbj);
    b1[j] = frag_b(pbj + 32);
  }
  v8f acc[4] = {};
#pragma unroll
  for (int j = 0; j < 4; ++j) acc[j] = WMMA_F16(a0, b0[j], acc[j]);
#pragma unroll
  for (int j = 0; j < 4; ++j) acc[j] = WMMA_F16(a1, b1[j], acc[j]);

  float* rowb = attn + ((long)bh * SS + q16 * 16) * SS;
  const int mr = (lane & 16) >> 1;
#pragma unroll
  for (int j = 0; j < 4; ++j) {
    const int kcol = ks * 64 + j * 16 + (lane & 15);
#pragma unroll
    for (int r = 0; r < 8; ++r)
      rowb[(long)(mr + r) * SS + kcol] = acc[j][r] * 0.125f;  // 64^-0.5
  }
}

// ---- in-place causal softmax over each length-2048 row --------------------
__global__ __launch_bounds__(256)
void softmax_kernel(float* __restrict__ attn) {
  __shared__ float red[256];
  const int row = blockIdx.x;             // bh*S + q
  const int q = row & (SS - 1);
  const int L = q + 1;                    // valid (k <= q)
  float* p = attn + (long)row * SS;
  const int tid = threadIdx.x;
  float mx = -1e30f;
  for (int i = tid; i < L; i += 256) mx = fmaxf(mx, p[i]);
  red[tid] = mx; __syncthreads();
  for (int s = 128; s > 0; s >>= 1) {
    if (tid < s) red[tid] = fmaxf(red[tid], red[tid + s]);
    __syncthreads();
  }
  mx = red[0]; __syncthreads();
  float sum = 0.f;
  for (int i = tid; i < L; i += 256) sum += __expf(p[i] - mx);
  red[tid] = sum; __syncthreads();
  for (int s = 128; s > 0; s >>= 1) {
    if (tid < s) red[tid] += red[tid + s];
    __syncthreads();
  }
  const float inv = 1.0f / red[0];
  for (int i = tid; i < SS; i += 256)
    p[i] = (i < L) ? __expf(p[i] - mx) * inv : 0.0f;   // exact zeros above diag
}

// ---- A-fragment built from f32 probabilities (convert in-register) --------
__device__ __forceinline__ v16h build_a_from_f32(const float* __restrict__ p0) {
  float4 f0 = *(const float4*)(p0);
  float4 f1 = *(const float4*)(p0 + 4);
  float4 f2 = *(const float4*)(p0 + 16);
  float4 f3 = *(const float4*)(p0 + 20);
  v16h a;
  a[0]  = (_Float16)f0.x; a[1]  = (_Float16)f0.y;
  a[2]  = (_Float16)f0.z; a[3]  = (_Float16)f0.w;
  a[4]  = (_Float16)f1.x; a[5]  = (_Float16)f1.y;
  a[6]  = (_Float16)f1.z; a[7]  = (_Float16)f1.w;
  a[8]  = (_Float16)f2.x; a[9]  = (_Float16)f2.y;
  a[10] = (_Float16)f2.z; a[11] = (_Float16)f2.w;
  a[12] = (_Float16)f3.x; a[13] = (_Float16)f3.y;
  a[14] = (_Float16)f3.z; a[15] = (_Float16)f3.w;
  return a;
}

// ---- ctx = P @ V  (ping-pong pipelined; K loop stops at diagonal block) ---
__global__ __launch_bounds__(32)
void attnv_kernel(const float* __restrict__ attn, const _Float16* __restrict__ Vt,
                  _Float16* __restrict__ ctx) {
  const int lane = threadIdx.x;
  const int q16  = blockIdx.x;        // 128
  const int hd16 = blockIdx.y;        // 4
  const int bh   = blockIdx.z;        // 32
  const float* Pb = attn + ((long)bh * SS + q16 * 16) * SS;
  const _Float16* Vb = Vt + (long)bh * HD * SS;
  const int off = (lane & 16) >> 1;
  const float* prow = Pb + (long)(lane & 15) * SS + off;
  const int vrow = hd16 * 16 + (lane & 15);
  const _Float16* pv = Vb + (long)vrow * SS + ((lane & 16) >> 1);
  const int n = (q16 * 16 + 16 + 31) >> 5;   // #K steps (probs 0 past diag), >=1

  v8f acc = {};
  v16h aA = build_a_from_f32(prow);
  v16h bA = frag_b(pv);
  int i = 1;
  for (; i + 1 < n; i += 2) {
    v16h aB = build_a_from_f32(prow + i * 32);
    v16h bB = frag_b(pv + i * 32);
    acc = WMMA_F16(aA, bA, acc);
    aA = build_a_from_f32(prow + (i + 1) * 32);
    bA = frag_b(pv + (i + 1) * 32);
    acc = WMMA_F16(aB, bB, acc);
  }
  if (i < n) {
    v16h aB = build_a_from_f32(prow + i * 32);
    v16h bB = frag_b(pv + i * 32);
    acc = WMMA_F16(aA, bA, acc);
    acc = WMMA_F16(aB, bB, acc);
  } else {
    acc = WMMA_F16(aA, bA, acc);
  }

  const int bb = bh >> 4, h = bh & 15;
  const int s0 = q16 * 16 + ((lane & 16) >> 1);
  const int d  = h * HD + hd16 * 16 + (lane & 15);
#pragma unroll
  for (int r = 0; r < 8; ++r)
    ctx[((long)bb * SS + (s0 + r)) * DD + d] = (_Float16)acc[r];
}

// ---------------------------------------------------------------------------
extern "C" void kernel_launch(void* const* d_in, const int* in_sizes, int n_in,
                              void* d_out, int out_size, void* d_ws, size_t ws_size,
                              hipStream_t stream) {
  const float* x  = (const float*)d_in[0];
  const float* Wq = (const float*)d_in[1];
  const float* bq = (const float*)d_in[2];
  const float* Wk = (const float*)d_in[3];
  const float* bk = (const float*)d_in[4];
  const float* Wv = (const float*)d_in[5];
  const float* bv = (const float*)d_in[6];
  const float* Wo = (const float*)d_in[7];
  const float* bo = (const float*)d_in[8];

  float* out  = (float*)d_out;                 // [B,S,D]
  float* attn = out + (long)BB * SS * DD;      // [B,H,S,S]

  // workspace layout (f16), total ~48 MB
  _Float16* ws   = (_Float16*)d_ws;
  _Float16* x_h  = ws;                                   // B*S*D
  _Float16* Wq_h = x_h  + (long)BB * SS * DD;            // D*D each
  _Float16* Wk_h = Wq_h + (long)DD * DD;
  _Float16* Wv_h = Wk_h + (long)DD * DD;
  _Float16* Wo_h = Wv_h + (long)DD * DD;
  _Float16* Q_h  = Wo_h + (long)DD * DD;                 // [b,h,s,hd]
  _Float16* K_h  = Q_h  + (long)BB * SS * DD;            // [b,h,s,hd]
  _Float16* Vt_h = K_h  + (long)BB * SS * DD;            // [b,h,hd,s]
  _Float16* ctx_h= Vt_h + (long)BB * SS * DD;            // [b,s,D]

  const int nx = BB * SS * DD, nw = DD * DD;
  f32_to_f16_kernel<<<(nx + 255) / 256, 256, 0, stream>>>(x,  x_h,  nx);
  f32_to_f16_kernel<<<(nw + 255) / 256, 256, 0, stream>>>(Wq, Wq_h, nw);
  f32_to_f16_kernel<<<(nw + 255) / 256, 256, 0, stream>>>(Wk, Wk_h, nw);
  f32_to_f16_kernel<<<(nw + 255) / 256, 256, 0, stream>>>(Wv, Wv_h, nw);
  f32_to_f16_kernel<<<(nw + 255) / 256, 256, 0, stream>>>(Wo, Wo_h, nw);

  dim3 gg(BB * SS / 16, DD / 64, 1);
  gemm_xwT_kernel<<<gg, 32, 0, stream>>>(x_h, Wq_h, bq, Q_h,  DD, 0);
  gemm_xwT_kernel<<<gg, 32, 0, stream>>>(x_h, Wk_h, bk, K_h,  DD, 0);
  gemm_xwT_kernel<<<gg, 32, 0, stream>>>(x_h, Wv_h, bv, Vt_h, DD, 1);

  dim3 gs(SS / 16, SS / 64, BB * HH);
  scores_kernel<<<gs, 32, 0, stream>>>(Q_h, K_h, attn);

  softmax_kernel<<<BB * HH * SS, 256, 0, stream>>>(attn);

  dim3 ga(SS / 16, HD / 16, BB * HH);
  attnv_kernel<<<ga, 32, 0, stream>>>(attn, Vt_h, ctx_h);

  gemm_xwT_kernel<<<gg, 32, 0, stream>>>(ctx_h, Wo_h, bo, d_out, DD, 2);
}
